// ProtoAutoEncoderMnist_33079838114337
// MI455X (gfx1250) — matrix operand order repaired and checked by reading, not compiled, not run on hardware
//
#include <hip/hip_runtime.h>
#include <math.h>

// ---------------------------------------------------------------------------
// CDNA5 (gfx1250) WMMA types / helpers
// ---------------------------------------------------------------------------
typedef __attribute__((ext_vector_type(16))) __bf16 v16bf;
typedef __attribute__((ext_vector_type(8)))  float  f32x8;
typedef __attribute__((ext_vector_type(4)))  unsigned int uint4v;

union Frag {
    v16bf v;
    unsigned short us[16];
    uint4v q[2];
};

__device__ __forceinline__ f32x8 zero8() {
    f32x8 z = {0.f, 0.f, 0.f, 0.f, 0.f, 0.f, 0.f, 0.f};
    return z;
}

__device__ __forceinline__ f32x8 wmma_bf16(v16bf a, v16bf b, f32x8 c) {
    // D = A(16x32 bf16) * B(32x16 bf16) + C(16x16 f32)
    return __builtin_amdgcn_wmma_f32_16x16x32_bf16(
        /*neg_a=*/false, a, /*neg_b=*/false, b,
        /*c_mod=*/(short)0, c, /*reuse_a=*/false, /*reuse_b=*/false);
}

// vectorized fragment load: 32 contiguous bytes (2 x b128)
__device__ __forceinline__ Frag load_frag(const unsigned short* p) {
    Frag f;
    const uint4v* q = (const uint4v*)p;
    f.q[0] = q[0];
    f.q[1] = q[1];
    return f;
}

// f32 -> bf16 (round-to-nearest-even), as raw bits
__device__ __forceinline__ unsigned short f2bf(float f) {
    unsigned int u = __float_as_uint(f);
    unsigned int r = u + 0x7fffu + ((u >> 16) & 1u);
    return (unsigned short)(r >> 16);
}
__device__ __forceinline__ float bf2f(unsigned short h) {
    return __uint_as_float(((unsigned int)h) << 16);
}

// A-matrix (16x32 bf16) element->K mapping, per ISA 7.12.2
// lane L holds row L&15; half h -> K = ((h>>3)<<4) + ((L>>4)<<3) + (h&7)
__device__ __forceinline__ int a_koff(int h, int lane) {
    return ((h >> 3) << 4) + ((lane >> 4) << 3) + (h & 7);
}
// B-matrix (32x16 bf16): lane L holds col L&15; contiguous K = ((L>>4)<<4) + h; unused here (B buffers use a_koff's order)
__device__ __forceinline__ int b_koff(int h, int lane) {
    return ((lane >> 4) << 4) + h;
}

// Swizzled layouts (fragment-major, 16 halves = 32B contiguous per lane):
//   A tile  : halfIdx = (((ks*NR + row)*2 + hi)<<4) + h,  K = ks*32 + ((h>>3)<<4) + (hi<<3) + (h&7)
//   B matrix: halfIdx = ((((ks*NT + nt)*2 + hi)*16 + l15)<<4) + h,  K = ks*32 + ((h>>3)<<4) + (hi<<3) + (h&7), n = nt*16+l15

// ---------------------------------------------------------------------------
// Problem constants
// ---------------------------------------------------------------------------
#define Bn 4096
#define Dd 256
#define Pp 512
#define BN_RS rsqrtf(1.f + 1e-5f)

// workspace layout (bytes); all offsets 32B-aligned for b128 access
static constexpr size_t OFF_H2  = 0;                                   // f32  B*16*49
static constexpr size_t OFF_W3  = OFF_H2  + (size_t)Bn * 784 * 4;      // bf16 160*256 (swizzled)
static constexpr size_t OFF_H3  = OFF_W3  + (size_t)160 * 256 * 2;     // bf16 B*25*256
static constexpr size_t OFF_PB  = OFF_H3  + (size_t)Bn * 25 * 256 * 2; // bf16 256*512 (swizzled)
static constexpr size_t OFF_P2  = OFF_PB  + (size_t)256 * 512 * 2;     // f32  512
static constexpr size_t OFF_BUP = OFF_P2  + (size_t)512 * 4;           // bf16 256*6400 (swizzled)
static constexpr size_t OFF_ZB  = OFF_BUP + (size_t)256 * 6400 * 2;    // bf16 B*256
static constexpr size_t OFF_Z5  = OFF_ZB  + (size_t)Bn * 256 * 2;      // bf16 B*6400
static constexpr size_t OFF_BD1 = OFF_Z5  + (size_t)Bn * 6400 * 2;     // bf16 2304*16 (swizzled)
static constexpr size_t OFF_H7  = OFF_BD1 + (size_t)2304 * 16 * 2;     // f32  B*16*49
static constexpr size_t OFF_H14 = OFF_H7  + (size_t)Bn * 784 * 4;      // f32  B*8*196

// ---------------------------------------------------------------------------
// K1: conv1 (1->8, s2 p1) + conv2 (8->16, s2 p1) + BN + relu, fused per image
// ---------------------------------------------------------------------------
__global__ __launch_bounds__(256) void enc_front(
    const float* __restrict__ x, const float* __restrict__ w1, const float* __restrict__ b1,
    const float* __restrict__ w2, const float* __restrict__ b2,
    const float* __restrict__ g2, const float* __restrict__ bt2,
    float* __restrict__ h2) {
    __shared__ float sx[784];
    __shared__ float sh1[8 * 196];
    int b = blockIdx.x, tid = threadIdx.x;
    for (int e = tid; e < 784; e += 256) sx[e] = x[(size_t)b * 784 + e];
    __syncthreads();
    for (int e = tid; e < 8 * 196; e += 256) {
        int c = e / 196, yx = e % 196, oy = yx / 14, ox = yx % 14;
        float acc = b1[c];
        for (int ky = 0; ky < 3; ++ky) {
            int iy = oy * 2 - 1 + ky;
            if (iy < 0 || iy >= 28) continue;
            for (int kx = 0; kx < 3; ++kx) {
                int ix = ox * 2 - 1 + kx;
                if (ix < 0 || ix >= 28) continue;
                acc += sx[iy * 28 + ix] * w1[c * 9 + ky * 3 + kx];
            }
        }
        sh1[e] = fmaxf(acc, 0.f);
    }
    __syncthreads();
    float rs = BN_RS;
    for (int e = tid; e < 16 * 49; e += 256) {
        int c = e / 49, yx = e % 49, oy = yx / 7, ox = yx % 7;
        float acc = b2[c];
        for (int ci = 0; ci < 8; ++ci)
            for (int ky = 0; ky < 3; ++ky) {
                int iy = oy * 2 - 1 + ky;
                if (iy < 0 || iy >= 14) continue;
                for (int kx = 0; kx < 3; ++kx) {
                    int ix = ox * 2 - 1 + kx;
                    if (ix < 0 || ix >= 14) continue;
                    acc += sh1[ci * 196 + iy * 14 + ix] * w2[c * 72 + ci * 9 + ky * 3 + kx];
                }
            }
        acc = acc * (g2[c] * rs) + bt2[c];
        h2[(size_t)b * 784 + e] = fmaxf(acc, 0.f);
    }
}

// ---------------------------------------------------------------------------
// K2: bf16 weight repack into WMMA fragment-major (swizzled) layouts
// ---------------------------------------------------------------------------
__global__ __launch_bounds__(256) void prep_kernel(
    const float* __restrict__ w3, const float* __restrict__ proto,
    const float* __restrict__ wup, const float* __restrict__ wd1,
    unsigned short* __restrict__ W3sw, unsigned short* __restrict__ Psw,
    float* __restrict__ p2, unsigned short* __restrict__ Bupsw,
    unsigned short* __restrict__ Bd1sw) {
    int tid = blockIdx.x * blockDim.x + threadIdx.x;
    int nth = gridDim.x * blockDim.x;
    // conv3 weights: K=144 (c*9+i*3+j) padded to 160, N=256, ks=5, NT=16
    for (int e = tid; e < 160 * 256; e += nth) {
        int h = e & 15, l15 = (e >> 4) & 15, hi = (e >> 8) & 1;
        int nt = (e >> 9) & 15, ks = e >> 13;
        int K = ks * 32 + ((h >> 3) << 4) + (hi << 3) + (h & 7), n = nt * 16 + l15;
        float v = (K < 144) ? w3[n * 144 + K] : 0.f;
        W3sw[e] = f2bf(v);
    }
    // proto^T: K=256, N=512, ks=8, NT=32
    for (int e = tid; e < 256 * 512; e += nth) {
        int h = e & 15, l15 = (e >> 4) & 15, hi = (e >> 8) & 1;
        int nt = (e >> 9) & 31, ks = e >> 14;
        int K = ks * 32 + ((h >> 3) << 4) + (hi << 3) + (h & 7), n = nt * 16 + l15;
        Psw[e] = f2bf(proto[n * 256 + K]);
    }
    for (int n = tid; n < 512; n += nth) {
        float s = 0.f;
        for (int k = 0; k < 256; ++k) { float v = proto[n * 256 + k]; s += v * v; }
        p2[n] = s;
    }
    // wup as GEMM B: K=256 (d), N=6400 (c*25+pos), ks=8, NT=400
    for (int e = tid; e < 256 * 6400; e += nth) {
        int h = e & 15, l15 = (e >> 4) & 15, hi = (e >> 8) & 1;
        int rem = e >> 9, nt = rem % 400, ks = rem / 400;
        int K = ks * 32 + ((h >> 3) << 4) + (hi << 3) + (h & 7), n = nt * 16 + l15;
        Bupsw[e] = f2bf(wup[(size_t)K * 6400 + n]);
    }
    // wd1 conv_t == conv of pad-2 input with flipped kernel; K=2304, N=16, ks=72
    for (int e = tid; e < 2304 * 16; e += nth) {
        int h = e & 15, c = (e >> 4) & 15, hi = (e >> 8) & 1, ks = e >> 9;
        int K = ks * 32 + ((h >> 3) << 4) + (hi << 3) + (h & 7);
        int d = K / 9, rr = K - 9 * d, i = rr / 3, j = rr - 3 * i;
        Bd1sw[e] = f2bf(wd1[d * 144 + c * 9 + (2 - i) * 3 + (2 - j)]);
    }
}

// ---------------------------------------------------------------------------
// K3: conv3 (16->256, 3x3, s1 p0) as WMMA implicit GEMM. M=25(pad32), K=160, N=256
// ---------------------------------------------------------------------------
__global__ __launch_bounds__(256) void conv3_wmma(
    const float* __restrict__ h2, const unsigned short* __restrict__ W3sw,
    const float* __restrict__ b3, unsigned short* __restrict__ h3bf) {
    __shared__ __align__(16) unsigned short sA[5 * 32 * 32]; // swizzled A, 5 ksteps x 32 rows
    int b = blockIdx.x, tid = threadIdx.x, lane = tid & 31, w = tid >> 5;
    for (int e = tid; e < 5 * 32 * 32; e += 256) {
        int h = e & 15, hi = (e >> 4) & 1, m = (e >> 5) & 31, ks = e >> 10;
        int K = ks * 32 + ((h >> 3) << 4) + (hi << 3) + (h & 7);
        float v = 0.f;
        if (m < 25 && K < 144) {
            int y = m / 5, xx = m % 5, c = K / 9, rr = K % 9, i = rr / 3, j = rr % 3;
            v = h2[(size_t)b * 784 + c * 49 + (y + i) * 7 + (xx + j)];
        }
        sA[e] = f2bf(v);
    }
    __syncthreads();
    f32x8 acc[2][2];
    acc[0][0] = zero8(); acc[0][1] = zero8(); acc[1][0] = zero8(); acc[1][1] = zero8();
    int nt0 = w * 2, l15 = lane & 15, hi = lane >> 4;
    for (int ks = 0; ks < 5; ++ks) {
        Frag a[2], bb[2];
        #pragma unroll
        for (int mt = 0; mt < 2; ++mt)
            a[mt] = load_frag(&sA[((ks * 32 + mt * 16 + l15) * 2 + hi) << 4]);
        #pragma unroll
        for (int q = 0; q < 2; ++q)
            bb[q] = load_frag(&W3sw[(((ks * 16 + nt0 + q) * 2 + hi) * 16 + l15) << 4]);
        #pragma unroll
        for (int mt = 0; mt < 2; ++mt)
            #pragma unroll
            for (int q = 0; q < 2; ++q)
                acc[mt][q] = wmma_bf16(a[mt].v, bb[q].v, acc[mt][q]);
    }
    #pragma unroll
    for (int mt = 0; mt < 2; ++mt)
        #pragma unroll
        for (int q = 0; q < 2; ++q) {
            int n = (nt0 + q) * 16 + l15;
            #pragma unroll
            for (int r = 0; r < 8; ++r) {
                int m = mt * 16 + r + (hi << 3);
                if (m < 25) {
                    float v = fmaxf(acc[mt][q][r] + b3[n], 0.f);
                    h3bf[((size_t)b * 25 + m) * 256 + n] = f2bf(v);
                }
            }
        }
}

// ---------------------------------------------------------------------------
// K4: xp = h*p^T via WMMA (25x256 x 256x512), distances, min, softmax, z = w@proto
// ---------------------------------------------------------------------------
__global__ __launch_bounds__(256) void proto_kernel(
    const unsigned short* __restrict__ h3bf, const unsigned short* __restrict__ Psw,
    const float* __restrict__ p2g, const float* __restrict__ proto,
    float* __restrict__ out_mind, unsigned short* __restrict__ zbf) {
    __shared__ __align__(16) unsigned short sA[8 * 32 * 32]; // swizzled A, 8 ksteps x 32 rows
    __shared__ float x2s[32];
    __shared__ float p2s[512];
    __shared__ float mins[512];
    __shared__ float red[256];
    __shared__ float wls[512];
    int b = blockIdx.x, tid = threadIdx.x, lane = tid & 31, w = tid >> 5;
    for (int e = tid; e < 8 * 32 * 32; e += 256) {
        int h = e & 15, hig = (e >> 4) & 1, m = (e >> 5) & 31, ks = e >> 10;
        int K = ks * 32 + ((h >> 3) << 4) + (hig << 3) + (h & 7);
        sA[e] = (m < 25) ? h3bf[((size_t)b * 25 + m) * 256 + K] : (unsigned short)0;
    }
    for (int e = tid; e < 512; e += 256) p2s[e] = p2g[e];
    if (tid < 32) {
        float s = 0.f;
        if (tid < 25) {
            const unsigned short* row = &h3bf[((size_t)b * 25 + tid) * 256];
            for (int d = 0; d < 256; ++d) { float v = bf2f(row[d]); s += v * v; }
        }
        x2s[tid] = s;
    }
    __syncthreads();
    f32x8 acc[2][4];
    #pragma unroll
    for (int mt = 0; mt < 2; ++mt)
        #pragma unroll
        for (int q = 0; q < 4; ++q) acc[mt][q] = zero8();
    int l15 = lane & 15, hi = lane >> 4;
    for (int ks = 0; ks < 8; ++ks) {
        Frag a[2];
        #pragma unroll
        for (int mt = 0; mt < 2; ++mt)
            a[mt] = load_frag(&sA[((ks * 32 + mt * 16 + l15) * 2 + hi) << 4]);
        #pragma unroll
        for (int q = 0; q < 4; ++q) {
            Frag bb = load_frag(&Psw[(((ks * 32 + w * 4 + q) * 2 + hi) * 16 + l15) << 4]);
            acc[0][q] = wmma_bf16(a[0].v, bb.v, acc[0][q]);
            acc[1][q] = wmma_bf16(a[1].v, bb.v, acc[1][q]);
        }
    }
    const float FINF = 3.0e38f;
    #pragma unroll
    for (int q = 0; q < 4; ++q) {
        int n = (w * 4 + q) * 16 + l15;
        float mn = FINF;
        #pragma unroll
        for (int mt = 0; mt < 2; ++mt)
            #pragma unroll
            for (int r = 0; r < 8; ++r) {
                int m = mt * 16 + r + (hi << 3);
                if (m < 25) {
                    float d2 = fmaxf(x2s[m] - 2.f * acc[mt][q][r] + p2s[n], 0.f);
                    mn = fminf(mn, d2);
                }
            }
        mn = fminf(mn, __shfl_xor(mn, 16, 32)); // combine the two lane halves
        mins[n] = mn;
    }
    __syncthreads();
    out_mind[(size_t)b * 512 + tid] = mins[tid];
    out_mind[(size_t)b * 512 + 256 + tid] = mins[tid + 256];
    // softmax over activations a = -min_d
    red[tid] = fminf(mins[tid], mins[tid + 256]);
    __syncthreads();
    for (int s = 128; s > 0; s >>= 1) {
        if (tid < s) red[tid] = fminf(red[tid], red[tid + s]);
        __syncthreads();
    }
    float mmin = red[0];
    wls[tid] = __expf(mmin - mins[tid]);
    wls[tid + 256] = __expf(mmin - mins[tid + 256]);
    __syncthreads();
    red[tid] = wls[tid] + wls[tid + 256];
    __syncthreads();
    for (int s = 128; s > 0; s >>= 1) {
        if (tid < s) red[tid] += red[tid + s];
        __syncthreads();
    }
    float inv = 1.f / red[0];
    // z[d] = sum_p w[p]*proto[p,d] (M=1 GEMV: VALU)
    float z = 0.f;
    for (int p = 0; p < 512; ++p) z += wls[p] * proto[p * 256 + tid];
    zbf[(size_t)b * 256 + tid] = f2bf(z * inv);
}

// ---------------------------------------------------------------------------
// K5: wup conv_t (1x1 spatial input -> pure GEMM): z5 = z(4096x256) @ Wup(256x6400)
// ---------------------------------------------------------------------------
__global__ __launch_bounds__(256) void zup_wmma(
    const unsigned short* __restrict__ zbf, const unsigned short* __restrict__ Bupsw,
    const float* __restrict__ bup, const float* __restrict__ gup,
    const float* __restrict__ btup, unsigned short* __restrict__ z5bf) {
    __shared__ __align__(16) unsigned short sA[8 * 16 * 32]; // swizzled A, 8 ksteps x 16 rows
    int mt = blockIdx.x, ng = blockIdx.y;
    int tid = threadIdx.x, lane = tid & 31, w = tid >> 5;
    for (int e = tid; e < 8 * 16 * 32; e += 256) {
        int h = e & 15, hig = (e >> 4) & 1, row = (e >> 5) & 15, ks = e >> 9;
        int K = ks * 32 + ((h >> 3) << 4) + (hig << 3) + (h & 7);
        sA[e] = zbf[((size_t)mt * 16 + row) * 256 + K];
    }
    __syncthreads();
    int nt = ng * 8 + w, l15 = lane & 15, hi = lane >> 4;
    int n = nt * 16 + l15;
    f32x8 acc = zero8();
    for (int ks = 0; ks < 8; ++ks) {
        Frag a = load_frag(&sA[((ks * 16 + l15) * 2 + hi) << 4]);
        const unsigned short* bp = &Bupsw[(size_t)((((ks * 400 + nt) * 2 + hi) * 16 + l15)) << 4];
        if (ks < 7)
            __builtin_prefetch(&Bupsw[(size_t)(((((ks + 1) * 400 + nt) * 2 + hi) * 16 + l15)) << 4], 0, 1);
        Frag bb = load_frag(bp);
        acc = wmma_bf16(a.v, bb.v, acc);
    }
    int c = n / 25;
    float sc = gup[c] * BN_RS, bi = bup[c], bt = btup[c];
    #pragma unroll
    for (int r = 0; r < 8; ++r) {
        int m = mt * 16 + r + (hi << 3);
        float v = fmaxf((acc[r] + bi) * sc + bt, 0.f);
        z5bf[(size_t)m * 6400 + n] = f2bf(v);
    }
}

// ---------------------------------------------------------------------------
// K6: wd1 conv_t (256->16, 3x3, s1 p0) as WMMA implicit GEMM on pad-2 z5.
//     per image: M=49(pad64), K=2304, N=16
// ---------------------------------------------------------------------------
__global__ __launch_bounds__(128) void wd1_wmma(
    const unsigned short* __restrict__ z5bf, const unsigned short* __restrict__ Bd1sw,
    const float* __restrict__ bd1, const float* __restrict__ gd1,
    const float* __restrict__ btd1, float* __restrict__ h7) {
    __shared__ unsigned short spad[256 * 81]; // z5 padded to 9x9, bf16
    int b = blockIdx.x, tid = threadIdx.x, lane = tid & 31, w = tid >> 5;
    for (int e = tid; e < 256 * 81; e += 128) spad[e] = 0;
    __syncthreads();
    for (int e = tid; e < 6400; e += 128) {
        int d = e / 25, pos = e % 25, y = pos / 5, x = pos % 5;
        spad[d * 81 + (y + 2) * 9 + (x + 2)] = z5bf[(size_t)b * 6400 + e];
    }
    __syncthreads();
    f32x8 acc = zero8();
    int l15 = lane & 15, hi = lane >> 4;
    int mrow = w * 16 + l15;
    bool mval = mrow < 49;
    int my = mval ? mrow / 7 : 0, mx = mval ? mrow % 7 : 0;
    for (int ks = 0; ks < 72; ++ks) {
        Frag a;
        #pragma unroll
        for (int h = 0; h < 16; ++h) {
            int k = ks * 32 + a_koff(h, lane);
            int d = k / 9, rr = k - 9 * d, i = rr / 3, j = rr - 3 * i;
            a.us[h] = mval ? spad[d * 81 + (my + i) * 9 + (mx + j)] : (unsigned short)0;
        }
        Frag bb = load_frag(&Bd1sw[((ks * 2 + hi) * 16 + l15) << 4]);
        acc = wmma_bf16(a.v, bb.v, acc);
    }
    int c = l15;
    float sc = gd1[c] * BN_RS, bi = bd1[c], bt = btd1[c];
    #pragma unroll
    for (int r = 0; r < 8; ++r) {
        int m = w * 16 + r + (hi << 3);
        if (m < 49) {
            float v = fmaxf((acc[r] + bi) * sc + bt, 0.f);
            h7[(size_t)b * 784 + c * 49 + m] = v;
        }
    }
}

// ---------------------------------------------------------------------------
// K7: wd2 conv_t (16->8, 3x3, s2 p1 op1): (16,7,7)->(8,14,14), BN+relu. VALU.
// ---------------------------------------------------------------------------
__global__ __launch_bounds__(256) void wd2_kernel(
    const float* __restrict__ h7, const float* __restrict__ wd2,
    const float* __restrict__ bd2, const float* __restrict__ gd2,
    const float* __restrict__ btd2, float* __restrict__ h14) {
    int idx = blockIdx.x * 256 + threadIdx.x;
    if (idx >= Bn * 8 * 196) return;
    int b = idx / 1568, rem = idx % 1568;
    int c = rem / 196, yx = rem % 196, y = yx / 14, x = yx % 14;
    float acc = 0.f;
    for (int ky = 0; ky < 3; ++ky) {
        int ty = y + 1 - ky;
        if (ty < 0 || (ty & 1)) continue;
        int iy = ty >> 1;
        if (iy >= 7) continue;
        for (int kx = 0; kx < 3; ++kx) {
            int tx = x + 1 - kx;
            if (tx < 0 || (tx & 1)) continue;
            int ix = tx >> 1;
            if (ix >= 7) continue;
            for (int d = 0; d < 16; ++d)
                acc += h7[(size_t)b * 784 + d * 49 + iy * 7 + ix] * wd2[d * 72 + c * 9 + ky * 3 + kx];
        }
    }
    float v = fmaxf((acc + bd2[c]) * (gd2[c] * BN_RS) + btd2[c], 0.f);
    h14[(size_t)b * 1568 + c * 196 + yx] = v;
}

// ---------------------------------------------------------------------------
// K8: wd3 conv_t (8->1, 3x3, s2 p1 op1): (8,14,14)->(1,28,28), sigmoid. VALU.
// ---------------------------------------------------------------------------
__global__ __launch_bounds__(256) void wd3_kernel(
    const float* __restrict__ h14, const float* __restrict__ wd3,
    const float* __restrict__ bd3, float* __restrict__ out) {
    int idx = blockIdx.x * 256 + threadIdx.x;
    if (idx >= Bn * 784) return;
    int b = idx / 784, yx = idx % 784, y = yx / 28, x = yx % 28;
    float acc = 0.f;
    for (int ky = 0; ky < 3; ++ky) {
        int ty = y + 1 - ky;
        if (ty < 0 || (ty & 1)) continue;
        int iy = ty >> 1;
        if (iy >= 14) continue;
        for (int kx = 0; kx < 3; ++kx) {
            int tx = x + 1 - kx;
            if (tx < 0 || (tx & 1)) continue;
            int ix = tx >> 1;
            if (ix >= 14) continue;
            for (int d = 0; d < 8; ++d)
                acc += h14[(size_t)b * 1568 + d * 196 + iy * 14 + ix] * wd3[d * 9 + ky * 3 + kx];
        }
    }
    float v = acc + bd3[0];
    out[(size_t)b * 784 + yx] = 1.f / (1.f + __expf(-v));
}

// ---------------------------------------------------------------------------
// launch
// ---------------------------------------------------------------------------
extern "C" void kernel_launch(void* const* d_in, const int* in_sizes, int n_in,
                              void* d_out, int out_size, void* d_ws, size_t ws_size,
                              hipStream_t stream) {
    const float* x     = (const float*)d_in[0];
    const float* w1    = (const float*)d_in[1];
    const float* b1    = (const float*)d_in[2];
    const float* w2    = (const float*)d_in[3];
    const float* b2    = (const float*)d_in[4];
    const float* g2    = (const float*)d_in[5];
    const float* bt2   = (const float*)d_in[6];
    const float* w3    = (const float*)d_in[7];
    const float* b3    = (const float*)d_in[8];
    const float* proto = (const float*)d_in[9];
    const float* wup   = (const float*)d_in[10];
    const float* bup   = (const float*)d_in[11];
    const float* gup   = (const float*)d_in[12];
    const float* btup  = (const float*)d_in[13];
    const float* wd1   = (const float*)d_in[14];
    const float* bd1   = (const float*)d_in[15];
    const float* gd1   = (const float*)d_in[16];
    const float* btd1  = (const float*)d_in[17];
    const float* wd2   = (const float*)d_in[18];
    const float* bd2   = (const float*)d_in[19];
    const float* gd2   = (const float*)d_in[20];
    const float* btd2  = (const float*)d_in[21];
    const float* wd3   = (const float*)d_in[22];
    const float* bd3   = (const float*)d_in[23];

    char* ws = (char*)d_ws;
    float*          h2    = (float*)(ws + OFF_H2);
    unsigned short* W3sw  = (unsigned short*)(ws + OFF_W3);
    unsigned short* h3bf  = (unsigned short*)(ws + OFF_H3);
    unsigned short* Psw   = (unsigned short*)(ws + OFF_PB);
    float*          p2    = (float*)(ws + OFF_P2);
    unsigned short* Bupsw = (unsigned short*)(ws + OFF_BUP);
    unsigned short* zbf   = (unsigned short*)(ws + OFF_ZB);
    unsigned short* z5bf  = (unsigned short*)(ws + OFF_Z5);
    unsigned short* Bd1sw = (unsigned short*)(ws + OFF_BD1);
    float*          h7    = (float*)(ws + OFF_H7);
    float*          h14   = (float*)(ws + OFF_H14);

    float* out      = (float*)d_out;                 // (B,1,28,28)
    float* out_mind = out + (size_t)Bn * 784;        // (B,512)

    enc_front<<<Bn, 256, 0, stream>>>(x, w1, b1, w2, b2, g2, bt2, h2);
    prep_kernel<<<64, 256, 0, stream>>>(w3, proto, wup, wd1, W3sw, Psw, p2, Bupsw, Bd1sw);
    conv3_wmma<<<Bn, 256, 0, stream>>>(h2, W3sw, b3, h3bf);
    proto_kernel<<<Bn, 256, 0, stream>>>(h3bf, Psw, p2, proto, out_mind, zbf);
    zup_wmma<<<dim3(Bn / 16, 50), 256, 0, stream>>>(zbf, Bupsw, bup, gup, btup, z5bf);
    wd1_wmma<<<Bn, 128, 0, stream>>>(z5bf, Bd1sw, bd1, gd1, btd1, h7);
    wd2_kernel<<<(Bn * 8 * 196 + 255) / 256, 256, 0, stream>>>(h7, wd2, bd2, gd2, btd2, h14);
    wd3_kernel<<<(Bn * 784 + 255) / 256, 256, 0, stream>>>(h14, wd3, bd3, out);
}
